// VGG_deformable_77489799954450
// MI455X (gfx1250) — compile-verified, matrix operand order checked by reading, not run on hardware
//
#include <hip/hip_runtime.h>
#include <hip/hip_bf16.h>

typedef __attribute__((ext_vector_type(16))) _Float16 v16h;
typedef __attribute__((ext_vector_type(8)))  float    v8f;

static __host__ __device__ inline int cdiv_i(int a, int b) { return (a + b - 1) / b; }

// CDNA5 16-bit WMMA fragment K-order within a 32-wide K-chunk:
// lane<16 reads K {0..7,16..23}, lane>=16 reads K {8..15,24..31}
// -> storage permutation = swap bits 3 and 4 of the K index.
static __device__ inline int swap34(int c) {
  return (c & ~24) | ((c & 8) << 1) | ((c & 16) >> 1);
}

// ---------------------------------------------------------------------------
// Pack fp32 weights [N, K] row-major into fragment-ordered f16 B:
//   out[((k>>5)*Np + n)*32 + swap34(k&31)], zero padded to [Kp, Np],
// Np a multiple of 64 so every GEMM tile is load-safe without guards.
// ---------------------------------------------------------------------------
__global__ __launch_bounds__(256) void pack_b_frag(const float* __restrict__ w,
                                                   _Float16* __restrict__ out,
                                                   int K, int N, int Kp, int Np) {
  long long i = (long long)blockIdx.x * 256 + threadIdx.x;
  if (i >= (long long)Kp * Np) return;
  int k = (int)(i / Np), n = (int)(i % Np);
  float v = (k < K && n < N) ? w[(size_t)n * K + k] : 0.f;
  out[((size_t)(k >> 5) * Np + n) * 32 + swap34(k & 31)] = (_Float16)v;
}

// Pack fp32 activations [M, K] row-major (K multiple of 32) into fragment-
// ordered f16 A: out[((k>>5)*M + m)*32 + swap34(k&31)].
__global__ __launch_bounds__(256) void pack_a_frag(const float* __restrict__ in,
                                                   _Float16* __restrict__ out,
                                                   int M, int K) {
  long long i = (long long)blockIdx.x * 256 + threadIdx.x;
  if (i >= (long long)M * K) return;
  int m = (int)(i / K), k = (int)(i % K);
  out[((size_t)(k >> 5) * M + m) * 32 + swap34(k & 31)] = (_Float16)in[(size_t)m * K + k];
}

// ---------------------------------------------------------------------------
// Deformable im2col, writing directly in WMMA fragment order:
//   acol[((k>>5)*M + m)*32 + swap34(k&31)] = bilinear(x[b,c], tap t)
// ---------------------------------------------------------------------------
__global__ __launch_bounds__(256) void deform_im2col(const float* __restrict__ x,
                                                     const float* __restrict__ off,
                                                     _Float16* __restrict__ acol,
                                                     int C, int H, int W, int Kp) {
  int HW = H * W;
  int M = 32 * HW;
  long long gid = (long long)blockIdx.x * 256 + threadIdx.x;
  if (gid >= (long long)M * Kp) return;
  int k = (int)(gid % Kp);
  int m = (int)(gid / Kp);
  size_t dst = ((size_t)(k >> 5) * M + m) * 32 + swap34(k & 31);
  if (k >= C * 9) { acol[dst] = (_Float16)0.f; return; }
  int c = k / 9, t = k % 9;
  int b = m / HW, p = m % HW;
  int yy = p / W, xx = p % W;
  int dy = t / 3 - 1, dx = t % 3 - 1;
  float oy = off[((size_t)(b * 18 + 2 * t)) * HW + p];
  float ox = off[((size_t)(b * 18 + 2 * t + 1)) * HW + p];
  float py = (float)yy + (float)dy + oy;
  float px = (float)xx + (float)dx + ox;
  float fy0 = floorf(py), fx0 = floorf(px);
  int iy0 = (int)fy0, ix0 = (int)fx0;
  float wy1 = py - fy0, wx1 = px - fx0;
  float wy0 = 1.f - wy1, wx0 = 1.f - wx1;

  const float* xb = x + (size_t)(b * C + c) * HW;
  float acc = 0.f;
#pragma unroll
  for (int j = 0; j < 4; ++j) {
    int ry = iy0 + (j >> 1);
    int rx = ix0 + (j & 1);
    bool valid = (ry >= 0) & (ry < H) & (rx >= 0) & (rx < W);
    int cy = ry < 0 ? 0 : (ry >= H ? H - 1 : ry);
    int cx = rx < 0 ? 0 : (rx >= W ? W - 1 : rx);
    float wgt = ((j >> 1) ? wy1 : wy0) * ((j & 1) ? wx1 : wx0);
    if (valid) acc += wgt * xb[cy * W + cx];
  }
  acol[dst] = (_Float16)acc;
}

// ---------------------------------------------------------------------------
// WMMA GEMM on fragment-ordered operands. No LDS, no barriers, no in-loop
// guards: each wave owns one 16x16 tile; per K-step each lane does two b128
// loads for A, two for B, then one v_wmma_f32_16x16x32_f16.
// Block = 8 waves = 32(M) x 64(N) macro tile. Np is a multiple of 64.
// A: [(Kp/32)][M][32] frag-order f16;  B: [(Kp/32)][Np][32] frag-order f16.
// ---------------------------------------------------------------------------
__global__ __launch_bounds__(256) void wmma_gemm(const _Float16* __restrict__ A,
                                                 const _Float16* __restrict__ Bm,
                                                 const float* __restrict__ bias,
                                                 float* __restrict__ D, int ldd,
                                                 int M, int N, int Np, int Ksteps,
                                                 int relu) {
  int tid  = threadIdx.x;
  int lane = tid & 31, wid = tid >> 5;
  int m0 = blockIdx.y * 32 + ((wid >> 2) << 4);   // this wave's 16-row tile
  int n0 = blockIdx.x * 64 + ((wid & 3) << 4);    // this wave's 16-col tile

  int mrow = m0 + (lane & 15);
  int nrow = n0 + (lane & 15);
  int hoff = (lane >= 16) ? 16 : 0;

  const _Float16* ap = A  + (size_t)mrow * 32 + hoff;
  const _Float16* bp = Bm + (size_t)nrow * 32 + hoff;
  size_t astep = (size_t)M * 32;
  size_t bstep = (size_t)Np * 32;

  v8f acc = {};
#pragma unroll 2
  for (int kc = 0; kc < Ksteps; ++kc) {
    v16h av = *(const v16h*)(ap);                  // 2x global_load_b128
    v16h bv = *(const v16h*)(bp);                  // 2x global_load_b128
    acc = __builtin_amdgcn_wmma_f32_16x16x32_f16(false, av, false, bv,
                                                 (short)0, acc, false, false);
    ap += astep;
    bp += bstep;
  }

  // C/D layout: VGPR j -> M = j + 8*(lane>=16); N = lane&15
  int n = n0 + (lane & 15);
  if (n < N) {
    float bias_v = bias ? bias[n] : 0.f;
#pragma unroll
    for (int j = 0; j < 8; ++j) {
      int m = m0 + j + ((lane >= 16) ? 8 : 0);
      float v = acc[j] + bias_v;
      if (relu) v = fmaxf(v, 0.f);
      D[(size_t)m * ldd + n] = v;
    }
  }
}

// ---------------------------------------------------------------------------
// BN batch statistics per channel (column of conv[M, Np]); stats[2n]=mean,
// stats[2n+1]=rsqrt(var+eps).
// ---------------------------------------------------------------------------
__global__ __launch_bounds__(256) void bn_stats(const float* __restrict__ conv,
                                                int M, int Np, float* __restrict__ stats) {
  int n = blockIdx.x;
  __shared__ float s1[256], s2[256];
  float a = 0.f, b = 0.f;
  for (int m = threadIdx.x; m < M; m += 256) {
    float v = conv[(size_t)m * Np + n];
    a += v; b += v * v;
  }
  s1[threadIdx.x] = a; s2[threadIdx.x] = b;
  __syncthreads();
  for (int s = 128; s > 0; s >>= 1) {
    if (threadIdx.x < s) { s1[threadIdx.x] += s1[threadIdx.x + s]; s2[threadIdx.x] += s2[threadIdx.x + s]; }
    __syncthreads();
  }
  if (threadIdx.x == 0) {
    float mean = s1[0] / (float)M;
    float var  = s2[0] / (float)M - mean * mean;
    stats[2 * n]     = mean;
    stats[2 * n + 1] = rsqrtf(var + 1e-5f);
  }
}

// Apply BN + ReLU and transpose [M=pixel, N=chan] -> NCHW fp32 for next layer.
__global__ __launch_bounds__(256) void bn_relu_nchw(const float* __restrict__ conv,
                                                    const float* __restrict__ stats,
                                                    const float* __restrict__ gamma,
                                                    const float* __restrict__ beta,
                                                    float* __restrict__ xout,
                                                    int N, int Np, int HW, int M) {
  long long i = (long long)blockIdx.x * 256 + threadIdx.x;
  if (i >= (long long)M * N) return;
  int m = (int)(i / N), n = (int)(i % N);
  int b = m / HW, p = m % HW;
  float v = conv[(size_t)m * Np + n];
  v = (v - stats[2 * n]) * stats[2 * n + 1] * gamma[n] + beta[n];
  v = fmaxf(v, 0.f);
  xout[(size_t)(b * N + n) * HW + p] = v;
}

// Permutation-indexed 2x2 max pool (gather by perm, pool, scatter by nperm).
__global__ __launch_bounds__(256) void maxpool_perm(const float* __restrict__ xin,
                                                    float* __restrict__ xout,
                                                    const int* __restrict__ perm,
                                                    const int* __restrict__ nperm,
                                                    int C, int H, int W) {
  int H2 = H >> 1, W2 = W >> 1, HW = H * W, HW2 = H2 * W2;
  long long i = (long long)blockIdx.x * 256 + threadIdx.x;
  if (i >= (long long)32 * C * HW2) return;
  int j  = (int)(i % HW2);
  int c  = (int)((i / HW2) % C);
  int b  = (int)(i / ((long long)C * HW2));
  int q  = nperm[j];
  int oy = q / W2, ox = q % W2;
  const float* xb = xin + (size_t)(b * C + c) * HW;
  float mx = -3.402823466e+38f;
#pragma unroll
  for (int d = 0; d < 4; ++d) {
    int src = perm[(2 * oy + (d >> 1)) * W + (2 * ox + (d & 1))];
    mx = fmaxf(mx, xb[src]);
  }
  xout[(size_t)(b * C + c) * HW2 + j] = mx;
}

// ---------------------------------------------------------------------------
// Host orchestration
// ---------------------------------------------------------------------------
extern "C" void kernel_launch(void* const* d_in, const int* in_sizes, int n_in,
                              void* d_out, int out_size, void* d_ws, size_t ws_size,
                              hipStream_t stream) {
  (void)in_sizes; (void)n_in; (void)out_size; (void)ws_size;

  static const int CI[13] = {3,64,64,128,128,256,256,256,512,512,512,512,512};
  static const int CO[13] = {64,64,128,128,256,256,256,512,512,512,512,512,512};
  static const int RS[13] = {32,32,16,16,8,8,8,4,4,4,2,2,2};
  static const int PL[13] = {-1,0,-1,1,-1,-1,2,-1,-1,3,-1,-1,4};

  const float* x0 = (const float*)d_in[0];
  const int* permp[5]; const int* npermp[5];
  for (int i = 0; i < 5; ++i) {
    permp[i]  = (const int*)d_in[66 + 2 * i];
    npermp[i] = (const int*)d_in[67 + 2 * i];
  }
  const float* fw[3] = {(const float*)d_in[76], (const float*)d_in[78], (const float*)d_in[80]};
  const float* fb[3] = {(const float*)d_in[77], (const float*)d_in[79], (const float*)d_in[81]};

  // Workspace carve
  char* ws = (char*)d_ws;
  float*    xbuf0 = (float*)(ws);                                    // 16 MB
  float*    xbuf1 = (float*)(ws + ((size_t)16 << 20));               // 16 MB
  _Float16* acol  = (_Float16*)(ws + ((size_t)32 << 20));            // 40 MB
  _Float16* wpk   = (_Float16*)(ws + ((size_t)72 << 20));            // 40 MB
  float*    conv  = (float*)(ws + ((size_t)112 << 20));              // 16 MB
  float*    stats = (float*)(ws + ((size_t)128 << 20));              // small
  _Float16* hbuf  = (_Float16*)(ws + ((size_t)128 << 20) + (1 << 16));
  float*    xbuf[2] = {xbuf0, xbuf1};

  const float* xin = x0;
  for (int L = 0; L < 13; ++L) {
    const float* offL = (const float*)d_in[1 + 5 * L + 0];
    const float* wL   = (const float*)d_in[1 + 5 * L + 1];
    const float* bL   = (const float*)d_in[1 + 5 * L + 2];
    const float* gL   = (const float*)d_in[1 + 5 * L + 3];
    const float* bbL  = (const float*)d_in[1 + 5 * L + 4];

    int Ci = CI[L], Co = CO[L], H = RS[L], W = H, HW = H * W;
    int M  = 32 * HW;
    int K  = Ci * 9;
    int Kp = cdiv_i(K, 32) * 32;
    int Np = cdiv_i(Co, 64) * 64;     // pad cols to macro-tile width: no guards

    long long bn = (long long)Kp * Np;
    pack_b_frag<<<(int)((bn + 255) / 256), 256, 0, stream>>>(wL, wpk, K, Co, Kp, Np);

    long long an = (long long)M * Kp;
    deform_im2col<<<(int)((an + 255) / 256), 256, 0, stream>>>(xin, offL, acol, Ci, H, W, Kp);

    dim3 gg(Np / 64, M / 32);
    wmma_gemm<<<gg, 256, 0, stream>>>(acol, wpk, bL, conv, Np, M, Co, Np, Kp / 32, 0);

    bn_stats<<<Co, 256, 0, stream>>>(conv, M, Np, stats);

    int bwrite = (xin == (const float*)xbuf[0]) ? 1 : 0;
    long long mn = (long long)M * Co;
    bn_relu_nchw<<<(int)((mn + 255) / 256), 256, 0, stream>>>(conv, stats, gL, bbL,
                                                              xbuf[bwrite], Co, Np, HW, M);
    if (PL[L] >= 0) {
      int H2 = H / 2;
      long long pn = (long long)32 * Co * H2 * H2;
      maxpool_perm<<<(int)((pn + 255) / 256), 256, 0, stream>>>(xbuf[bwrite], xbuf[1 - bwrite],
                                                                permp[PL[L]], npermp[PL[L]],
                                                                Co, H, W);
      xin = xbuf[1 - bwrite];
    } else {
      xin = xbuf[bwrite];
    }
  }

  // FC head: h = flatten(x) -> [32, 512] fragment-packed
  pack_a_frag<<<cdiv_i(32 * 512, 256), 256, 0, stream>>>(xin, hbuf, 32, 512);

  // FC1: 512 -> 4096, ReLU
  pack_b_frag<<<cdiv_i(512 * 4096, 256), 256, 0, stream>>>(fw[0], wpk, 512, 4096, 512, 4096);
  wmma_gemm<<<dim3(4096 / 64, 1), 256, 0, stream>>>(hbuf, wpk, fb[0], conv, 4096,
                                                    32, 4096, 4096, 512 / 32, 1);
  pack_a_frag<<<cdiv_i(32 * 4096, 256), 256, 0, stream>>>(conv, hbuf, 32, 4096);

  // FC2: 4096 -> 4096, ReLU
  pack_b_frag<<<cdiv_i(4096 * 4096, 256), 256, 0, stream>>>(fw[1], wpk, 4096, 4096, 4096, 4096);
  wmma_gemm<<<dim3(4096 / 64, 1), 256, 0, stream>>>(hbuf, wpk, fb[1], conv, 4096,
                                                    32, 4096, 4096, 4096 / 32, 1);
  pack_a_frag<<<cdiv_i(32 * 4096, 256), 256, 0, stream>>>(conv, hbuf, 32, 4096);

  // FC3: 4096 -> 100, no ReLU, write d_out directly (Np padded to 128)
  pack_b_frag<<<cdiv_i(4096 * 128, 256), 256, 0, stream>>>(fw[2], wpk, 4096, 100, 4096, 128);
  wmma_gemm<<<dim3(128 / 64, 1), 256, 0, stream>>>(hbuf, wpk, fb[2], (float*)d_out, 100,
                                                   32, 100, 128, 4096 / 32, 0);
}
